// gnn_model_29231547417227
// MI455X (gfx1250) — matrix operand order repaired and checked by reading, not compiled, not run on hardware
//
#include <hip/hip_runtime.h>

typedef __bf16 bf16;
typedef __attribute__((ext_vector_type(16))) __bf16 v16bf;
typedef __attribute__((ext_vector_type(8)))  __bf16 v8bf;
typedef __attribute__((ext_vector_type(8)))  float  v8f;

#define HD 256          // hidden width
#define FLD 1280        // feature buffer leading dim (5*H)
#define BN_EPS 1e-3f

// ---------------------------------------------------------------------------
// Weight prep: fp32 (K x 256) row-major -> bf16 hi/lo in WMMA B-fragment order
// Layout: [ntile(16)][kstep(K/32)][lane(32)][j(16)], lane=(kk&16)|(n&15), j=kk&15
// ---------------------------------------------------------------------------
__global__ void prep_w_kernel(const float* __restrict__ W,
                              bf16* __restrict__ whi, bf16* __restrict__ wlo, int K) {
    int idx = blockIdx.x * 256 + threadIdx.x;
    if (idx >= K * 256) return;
    int k = idx >> 8, n = idx & 255;
    float x = W[idx];
    bf16 h = (bf16)x;
    bf16 l = (bf16)(x - (float)h);
    int nk = K >> 5;
    int nt = n >> 4, ks = k >> 5, kk = k & 31;
    int lane = (kk & 16) | (n & 15);
    int j = kk & 15;
    size_t off = (((size_t)(nt * nk + ks)) * 32 + lane) * 16 + j;
    whi[off] = h;
    wlo[off] = l;
}

// ---------------------------------------------------------------------------
// Fused GEMM (M x K @ K x 256) + bias + BatchNorm + PReLU via bf16x3 WMMA.
// Block: 512 thr = 16 waves (4 along M x 4 along N); block tile 128 x 256;
// wave tile 32 x 64 (2 m-tiles x 4 n-tiles). A is loaded fp32 from global,
// split to bf16 hi/lo ONCE per block by a cooperative loader that writes LDS
// directly in WMMA A-fragment order; waves read fragments with ds_load_b128.
// Grid: (ceil(M/128), 1).
// ---------------------------------------------------------------------------
__global__ __launch_bounds__(512)
void gemm_bn_prelu_kernel(const float* __restrict__ A, long lda,
                          const bf16* __restrict__ Whi, const bf16* __restrict__ Wlo,
                          const float* __restrict__ bias, const float* __restrict__ bnp,
                          const float* __restrict__ alpha,
                          float* __restrict__ out, long ldo, long col_off,
                          int M, int K) {
    __shared__ bf16 sAhi[128 * 32];   // [mt(8)][lane(32)][j(16)]  8 KB
    __shared__ bf16 sAlo[128 * 32];   //                            8 KB

    const int tid  = threadIdx.x;
    const int lane = tid & 31;
    const int wid  = tid >> 5;
    const int wm   = wid & 3;   // 4 waves along M (32 rows each)
    const int wn   = wid >> 2;  // 4 waves along N (64 cols each)
    const int rowbase = blockIdx.x * 128 + wm * 32;
    const int colbase = wn * 64;
    const int nk   = K >> 5;
    const int mlo  = lane & 15;

    // ---- cooperative A loader mapping (8 contiguous k per thread) ----
    const int lrow = tid >> 2;                       // 0..127 block-local row
    const int kseg = tid & 3;                        // 0..3 (8-k segment)
    const size_t growoff = (size_t)min(blockIdx.x * 128 + lrow, M - 1) * (size_t)lda;
    // fragment-order dest: kseg&2 -> lane-half, kseg&1 -> j-half (contiguous 8)
    const int dmt   = lrow >> 4;
    const int dlane = ((kseg & 2) << 3) | (lrow & 15);
    const int djb   = (kseg & 1) ? 8 : 0;
    bf16* dsth = sAhi + ((dmt * 32 + dlane) * 16 + djb);
    bf16* dstl = sAlo + ((dmt * 32 + dlane) * 16 + djb);

    // ---- per-wave fragment read addresses ----
    const int mt0 = wm * 2, mt1 = wm * 2 + 1;
    const bf16* fAh0 = sAhi + (mt0 * 32 + lane) * 16;
    const bf16* fAh1 = sAhi + (mt1 * 32 + lane) * 16;
    const bf16* fAl0 = sAlo + (mt0 * 32 + lane) * 16;
    const bf16* fAl1 = sAlo + (mt1 * 32 + lane) * 16;

    v8f acc[2][4] = {};

    for (int ks = 0; ks < nk; ++ks) {
        // ---- cooperative global load + bf16 hi/lo split ----
        const float4* p = reinterpret_cast<const float4*>(A + growoff + ks * 32 + kseg * 8);
        float4 u0 = p[0], u1 = p[1];
        float xs[8] = {u0.x, u0.y, u0.z, u0.w, u1.x, u1.y, u1.z, u1.w};
        v8bf hv, lv;
#pragma unroll
        for (int j = 0; j < 8; ++j) {
            bf16 h = (bf16)xs[j];
            hv[j] = h;
            lv[j] = (bf16)(xs[j] - (float)h);
        }
        __syncthreads();                       // prev-iter LDS reads complete
        *reinterpret_cast<v8bf*>(dsth) = hv;   // single 16B ds_store each
        *reinterpret_cast<v8bf*>(dstl) = lv;
        __syncthreads();                       // stores visible to all waves

        // ---- read A fragments (ds_load_b128 pairs, broadcast to 4 waves) ----
        v16bf a0h = *reinterpret_cast<const v16bf*>(fAh0);
        v16bf a1h = *reinterpret_cast<const v16bf*>(fAh1);
        v16bf a0l = *reinterpret_cast<const v16bf*>(fAl0);
        v16bf a1l = *reinterpret_cast<const v16bf*>(fAl1);

        // ---- per n-tile: load B fragments (L2-hot), 6 WMMAs (bf16x3) ----
#pragma unroll
        for (int ni = 0; ni < 4; ++ni) {
            const int ntg = wn * 4 + ni;
            const size_t bo = ((size_t)ntg * nk + ks) * 512 + (size_t)lane * 16;
            v16bf bh = *reinterpret_cast<const v16bf*>(Whi + bo);
            v16bf bl = *reinterpret_cast<const v16bf*>(Wlo + bo);

            acc[0][ni] = __builtin_amdgcn_wmma_f32_16x16x32_bf16(false, a0h, false, bh, (short)0, acc[0][ni], false, false);
            acc[0][ni] = __builtin_amdgcn_wmma_f32_16x16x32_bf16(false, a0h, false, bl, (short)0, acc[0][ni], false, false);
            acc[0][ni] = __builtin_amdgcn_wmma_f32_16x16x32_bf16(false, a0l, false, bh, (short)0, acc[0][ni], false, false);

            acc[1][ni] = __builtin_amdgcn_wmma_f32_16x16x32_bf16(false, a1h, false, bh, (short)0, acc[1][ni], false, false);
            acc[1][ni] = __builtin_amdgcn_wmma_f32_16x16x32_bf16(false, a1h, false, bl, (short)0, acc[1][ni], false, false);
            acc[1][ni] = __builtin_amdgcn_wmma_f32_16x16x32_bf16(false, a1l, false, bh, (short)0, acc[1][ni], false, false);
        }
    }

    // Epilogue: bias + BN + PReLU. D layout: row = r + 8*(lane>=16), col = lane&15.
    const int rhalf = (lane >> 4) << 3;
#pragma unroll
    for (int mi = 0; mi < 2; ++mi) {
        const int rowb = rowbase + mi * 16 + rhalf;
#pragma unroll
        for (int ni = 0; ni < 4; ++ni) {
            const int col = colbase + ni * 16 + mlo;
            const float bia = bias[col];
            const float g  = bnp[col];
            const float be = bnp[HD + col];
            const float mu = bnp[2 * HD + col];
            const float va = bnp[3 * HD + col];
            const float sc = g * rsqrtf(va + BN_EPS);
            const float al = alpha[col];
#pragma unroll
            for (int r = 0; r < 8; ++r) {
                const int row = rowb + r;
                if (row < M) {
                    float xv = acc[mi][ni][r] + bia;
                    float y  = sc * (xv - mu) + be;
                    y = (y >= 0.f) ? y : al * y;
                    out[(size_t)row * ldo + col_off + col] = y;
                }
            }
        }
    }
}

// ---------------------------------------------------------------------------
// Zero a 256-wide column slab of the feature buffer.
// ---------------------------------------------------------------------------
__global__ void zero_slab_kernel(float* __restrict__ feat, int col_off, int n) {
    int t = blockIdx.x * 256 + threadIdx.x;
    if (t < n) {
        int row = t >> 8, c = t & 255;
        feat[(size_t)row * FLD + col_off + c] = 0.f;
    }
}

// ---------------------------------------------------------------------------
// Edge scatter: feat[tgt, col_off + :] += h[src, :]   (float4 grain, f32 atomics)
// ---------------------------------------------------------------------------
__global__ void scatter_edges_kernel(const int* __restrict__ ei, const float* __restrict__ h,
                                     float* __restrict__ feat, int col_off, int E) {
    int t = blockIdx.x * 256 + threadIdx.x;
    if (t >= E * 64) return;
    int e = t >> 6;
    int q = (t & 63) << 2;
    int tgt = ei[2 * e];
    int src = ei[2 * e + 1];
    float4 v = *reinterpret_cast<const float4*>(h + (size_t)src * HD + q);
    float* d = feat + (size_t)tgt * FLD + col_off + q;
    atomicAdd(d + 0, v.x);
    atomicAdd(d + 1, v.y);
    atomicAdd(d + 2, v.z);
    atomicAdd(d + 3, v.w);
}

// ---------------------------------------------------------------------------
// post1 (256 -> 1) + BN(1ch) + segment_sum over graphs. One wave per node.
// ---------------------------------------------------------------------------
__global__ __launch_bounds__(256)
void post1_seg_kernel(const float* __restrict__ h, const float* __restrict__ W,
                      const float* __restrict__ b, const float* __restrict__ bnp,
                      const int* __restrict__ seg, float* __restrict__ out, int M) {
    int lane = threadIdx.x & 31;
    int wid  = threadIdx.x >> 5;
    int node = blockIdx.x * 8 + wid;
    if (node >= M) return;
    const float* hr = h + (size_t)node * HD;
    float s = 0.f;
#pragma unroll
    for (int t = 0; t < 8; ++t) s += hr[lane + 32 * t] * W[lane + 32 * t];
#pragma unroll
    for (int o = 16; o >= 1; o >>= 1) s += __shfl_xor(s, o, 32);
    if (lane == 0) {
        float x = s + b[0];
        float y = bnp[0] * (x - bnp[2]) * rsqrtf(bnp[3] + BN_EPS) + bnp[1];
        atomicAdd(out + seg[node], y);
    }
}

__global__ void zero_out_kernel(float* __restrict__ out, int n) {
    int t = blockIdx.x * 256 + threadIdx.x;
    if (t < n) out[t] = 0.f;
}

// ---------------------------------------------------------------------------
extern "C" void kernel_launch(void* const* d_in, const int* in_sizes, int n_in,
                              void* d_out, int out_size, void* d_ws, size_t ws_size,
                              hipStream_t stream) {
    const float* x   = (const float*)d_in[0];
    const int*   ei  = (const int*)d_in[1];
    const int*   seg = (const int*)d_in[2];
    const int N = in_sizes[0] / 128;  // F = 128
    const int E = in_sizes[1] / 2;

    // Layer params: pre0, pre1, g0..g3, post0 at 4 + 4*i; post1 at 32..34.
    const float *Wp[8], *Bp[8], *BNp[8], *Ap[7];
    for (int i = 0; i < 7; ++i) {
        Wp[i]  = (const float*)d_in[4 + 4 * i];
        Bp[i]  = (const float*)d_in[5 + 4 * i];
        BNp[i] = (const float*)d_in[6 + 4 * i];
        Ap[i]  = (const float*)d_in[7 + 4 * i];
    }
    Wp[7]  = (const float*)d_in[32];
    Bp[7]  = (const float*)d_in[33];
    BNp[7] = (const float*)d_in[34];

    // Workspace layout: feat (N x 1280 f32) | h (N x 256 f32) | bf16 weights
    char*  ws   = (char*)d_ws;
    float* feat = (float*)ws;
    float* h    = (float*)(ws + (size_t)N * FLD * 4);
    char*  wfp  = ws + (size_t)N * FLD * 4 + (size_t)N * HD * 4;

    const int Ks[7] = {128, 256, 256, 512, 768, 1024, 1280};
    bf16 *Whi[7], *Wlo[7];
    {
        size_t off = 0;
        for (int i = 0; i < 7; ++i) {
            Whi[i] = (bf16*)(wfp + off);
            Wlo[i] = (bf16*)(wfp + off + (size_t)Ks[i] * 256 * 2);
            off += (size_t)Ks[i] * 256 * 4;  // hi + lo, 2B each
        }
    }

    dim3 blk(256);
    for (int i = 0; i < 7; ++i) {
        int n = Ks[i] * 256;
        prep_w_kernel<<<(n + 255) / 256, blk, 0, stream>>>(Wp[i], Whi[i], Wlo[i], Ks[i]);
    }

    dim3 gblk(512);
    dim3 ggrid((N + 127) / 128);

    // pre0: x (K=128) -> h
    gemm_bn_prelu_kernel<<<ggrid, gblk, 0, stream>>>(x, 128, Whi[0], Wlo[0], Bp[0], BNp[0], Ap[0],
                                                     h, HD, 0, N, 128);
    // pre1: h (K=256) -> feat cols [1024:1280)
    gemm_bn_prelu_kernel<<<ggrid, gblk, 0, stream>>>(h, HD, Whi[1], Wlo[1], Bp[1], BNp[1], Ap[1],
                                                     feat, FLD, 1024, N, 256);

    // Graph conv layers: input = contiguous column suffix, z-slab prepended.
    const int colz[4] = {768, 512, 256, 0};
    const int kg[4]   = {256, 512, 768, 1024};
    for (int g = 0; g < 4; ++g) {
        const int incol = colz[g] + 256;
        gemm_bn_prelu_kernel<<<ggrid, gblk, 0, stream>>>(feat + incol, FLD,
                                                         Whi[2 + g], Wlo[2 + g],
                                                         Bp[2 + g], BNp[2 + g], Ap[2 + g],
                                                         h, HD, 0, N, kg[g]);
        int nz = N * 256;
        zero_slab_kernel<<<(nz + 255) / 256, blk, 0, stream>>>(feat, colz[g], nz);
        int ne = E * 64;
        scatter_edges_kernel<<<(ne + 255) / 256, blk, 0, stream>>>(ei, h, feat, colz[g], E);
    }

    // post0: feat (K=1280) -> h
    gemm_bn_prelu_kernel<<<ggrid, gblk, 0, stream>>>(feat, FLD, Whi[6], Wlo[6],
                                                     Bp[6], BNp[6], Ap[6],
                                                     h, HD, 0, N, 1280);

    // post1 + BN + graph segment-sum
    zero_out_kernel<<<(out_size + 255) / 256, blk, 0, stream>>>((float*)d_out, out_size);
    post1_seg_kernel<<<(N + 7) / 8, blk, 0, stream>>>(h, Wp[7], Bp[7], BNp[7], seg,
                                                      (float*)d_out, N);
}
